// MultiHeadSelfAttention_87892210745505
// MI455X (gfx1250) — matrix operand order repaired and checked by reading, not compiled, not run on hardware
//
#include <hip/hip_runtime.h>
#include <hip/hip_bf16.h>

typedef __attribute__((ext_vector_type(16))) __bf16 v16bf;
typedef __attribute__((ext_vector_type(8)))  __bf16 v8bf;
typedef __attribute__((ext_vector_type(8)))  float  v8f;
typedef __attribute__((ext_vector_type(4)))  int    v4i;

#define D_MODEL 2048
#define SEQ     2048
#define BATCH   2
#define NHEAD   16
#define DK      128
#define M_TOT   (BATCH * SEQ)          // 4096 rows of (b,s)

static __device__ __forceinline__ v8f wmma_bf16(v16bf a, v16bf b, v8f c) {
  // D = A(16x32) x B(32x16) + C, f32 accumulate
  return __builtin_amdgcn_wmma_f32_16x16x32_bf16(false, a, false, b, (short)0, c, false, false);
}

static __device__ __forceinline__ v8f vzero8() {
  v8f z = {0.f, 0.f, 0.f, 0.f, 0.f, 0.f, 0.f, 0.f};
  return z;
}

// ---- CDNA5 async global->LDS copy (ASYNCcnt path) -------------------------
static __device__ __forceinline__ void async_g2l_b128(const __bf16* g, uint32_t lds_off) {
#if __has_builtin(__builtin_amdgcn_global_load_async_to_lds_b128)
  typedef __attribute__((address_space(1))) v4i* gp1_t;   // global int4*
  typedef __attribute__((address_space(3))) v4i* lp3_t;   // LDS int4*
  __builtin_amdgcn_global_load_async_to_lds_b128((gp1_t)(uintptr_t)g,
                                                 (lp3_t)(uintptr_t)lds_off, 0, 0);
#else
  unsigned long long ga = (unsigned long long)(uintptr_t)g;
  asm volatile("global_load_async_to_lds_b128 %0, %1, off"
               :: "v"(lds_off), "v"(ga) : "memory");
#endif
}

static __device__ __forceinline__ void wait_async0() {
#if __has_builtin(__builtin_amdgcn_s_wait_asynccnt)
  __builtin_amdgcn_s_wait_asynccnt(0);
#else
  asm volatile("s_wait_asynccnt 0" ::: "memory");
#endif
}

static __device__ __forceinline__ uint32_t lds_off_of(const void* p) {
  return (uint32_t)(uintptr_t)p;   // generic->LDS aperture: low 32 bits = LDS offset
}

// ---------------------------------------------------------------------------
// Kernel 1: f32 -> bf16 cast
// ---------------------------------------------------------------------------
__global__ void cast_f32_bf16(const float* __restrict__ in, __bf16* __restrict__ out, int n) {
  int i = blockIdx.x * blockDim.x + threadIdx.x;
  if (i < n) out[i] = (__bf16)in[i];
}

// ---------------------------------------------------------------------------
// Kernel 2: fused QKV projection (NT GEMM) + RoPE on Q/K, V stored transposed.
// All 8 waves of a block share one 16-row A strip -> async-stage A into LDS.
// ---------------------------------------------------------------------------
__global__ void qkv_rope_kernel(const __bf16* __restrict__ xb,
                                const __bf16* __restrict__ wqb,
                                const __bf16* __restrict__ wkb,
                                const __bf16* __restrict__ wvb,
                                __bf16* __restrict__ Qr,
                                __bf16* __restrict__ Kr,
                                __bf16* __restrict__ Vt) {
  __shared__ __align__(16) __bf16 As[16][136];   // 16 rows x 128 K (+8 pad)

  const int tid  = threadIdx.x;
  const int lane = tid & 31;
  const int wv_id = tid >> 5;
  const int l15 = lane & 15;
  const int hi  = lane >> 4;

  const int NT = D_MODEL / 64;   // 32
  const int MT = M_TOT / 16;     // 256
  int task  = blockIdx.x * 8 + wv_id;          // 3*256*32 tasks; block shares mtile/mat
  int ntile = task % NT;
  int mtile = (task / NT) % MT;
  int mat   = task / (NT * MT);                // 0=Q 1=K 2=V

  const __bf16* W = (mat == 0) ? wqb : (mat == 1) ? wkb : wvb;
  size_t m0 = (size_t)mtile * 16;
  int n0 = ntile * 64;

  // per-thread staging slice: row = tid/16, 8-elt chunk = (tid%16)*8
  const int arow_i = tid >> 4;
  const int achk   = (tid & 15) * 8;
  const __bf16* asrc = xb + (m0 + (size_t)arow_i) * D_MODEL + achk;
  const uint32_t adst = lds_off_of(&As[arow_i][achk]);

  v8f acc[4];
  #pragma unroll
  for (int j = 0; j < 4; ++j) acc[j] = vzero8();

  for (int k0 = 0; k0 < D_MODEL; k0 += 128) {
    async_g2l_b128(asrc + k0, adst);
    wait_async0();
    __syncthreads();
    #pragma unroll
    for (int kk = 0; kk < 128; kk += 32) {
      union { v16bf v; v8bf h[2]; } a;
      a.h[0] = *(const v8bf*)&As[l15][kk + 16 * hi];
      a.h[1] = *(const v8bf*)&As[l15][kk + 16 * hi + 8];
      #pragma unroll
      for (int j = 0; j < 4; ++j) {
        v16bf b = *(const v16bf*)(W + (size_t)(n0 + 16 * j + l15) * D_MODEL + k0 + kk + 16 * hi);
        acc[j] = wmma_bf16(a.v, b, acc[j]);
      }
    }
    __syncthreads();
  }

  const int bb = (int)(m0 >> 11);       // / SEQ
  const int s0 = (int)(m0 & (SEQ - 1));

  if (mat == 2) {
    // V: store transposed [B,H,dk,S]; C-frag rows (s) are contiguous per lane.
    #pragma unroll
    for (int j = 0; j < 4; ++j) {
      int n = n0 + 16 * j + l15;
      int h = n >> 7, vd = n & 127;
      v8bf pk;
      #pragma unroll
      for (int r = 0; r < 8; ++r) pk[r] = (__bf16)acc[j][r];
      __bf16* dst = Vt + ((size_t)(bb * NHEAD + h) * DK + vd) * SEQ + s0 + 8 * hi;
      *(v8bf*)dst = pk;
    }
  } else {
    __bf16* Out = (mat == 0) ? Qr : Kr;
    #pragma unroll
    for (int j = 0; j < 4; ++j) {
      int n = n0 + 16 * j + l15;
      int h = n >> 7, vd = n & 127;
      float inv = __powf(10000.0f, -(float)(vd & ~1) * (1.0f / 128.0f));
      #pragma unroll
      for (int r = 0; r < 8; ++r) {
        int s = s0 + 8 * hi + r;
        float c = acc[j][r];
        float partner = __shfl_xor(c, 1, 32);
        float sn, cs;
        __sincosf((float)s * inv, &sn, &cs);
        float o = (vd & 1) ? fmaf(c, cs, partner * sn) : fmaf(c, cs, -partner * sn);
        Out[((size_t)(bb * NHEAD + h) * SEQ + s) * DK + vd] = (__bf16)o;
      }
    }
  }
}

// ---------------------------------------------------------------------------
// Kernel 3: causal flash attention, block-cooperative K/V staging.
// Block = 8 waves, same (b,h), 8 consecutive 16-query strips (128 q rows).
// Per s-step: async-stage K(32x128) and Vt(128x32) into LDS once per block.
// ---------------------------------------------------------------------------
__global__ void attn_kernel(const __bf16* __restrict__ Qr,
                            const __bf16* __restrict__ Kr,
                            const __bf16* __restrict__ Vt,
                            __bf16* __restrict__ Ao) {
  __shared__ __align__(16) __bf16 Ks[32][136];      // 32 s-rows x 128 dk (+pad)
  __shared__ __align__(16) __bf16 Vs[128][40];      // 128 vdims x 32 s (+pad)
  __shared__ __align__(64) __bf16 Pbuf[8][16][32];  // per-wave P^T tile

  const int tid  = threadIdx.x;
  const int lane = tid & 31;
  const int wv_id = tid >> 5;
  const int l15 = lane & 15;
  const int hi  = lane >> 4;

  int task  = blockIdx.x * 8 + wv_id;      // 2*16*128 tasks
  int qtile = task & 127;                  // SEQ/16 = 128 per bh (8 | 128)
  int bh    = task >> 7;                   // b*NHEAD + h (same for whole block)

  size_t base  = (size_t)bh * SEQ * DK;    // Q/K per (b,h)
  size_t vbase = (size_t)bh * DK * SEQ;    // Vt per (b,h)
  int q0 = qtile * 16;
  int q0blk = ((blockIdx.x * 8) & 127) * 16;   // first q row of block
  int smax = q0blk + 127;                      // last q row in block

  // Q fragments (B operand of scoresT): lane = q column, 4 chunks of dk=32
  v16bf qf[4];
  const __bf16* qrow = Qr + base + (size_t)(q0 + l15) * DK;
  #pragma unroll
  for (int d = 0; d < 4; ++d) qf[d] = *(const v16bf*)(qrow + 32 * d + 16 * hi);

  // staging slices: K = 512 b128-chunks, V = 512 b128-chunks, 2+2 per thread
  const int kr0 = tid >> 4,          ko0 = (tid & 15) * 8;
  const int kr1 = (tid + 256) >> 4,  ko1 = ((tid + 256) & 15) * 8;
  const int vr0 = tid >> 2,          vo0 = (tid & 3) * 8;
  const int vr1 = (tid + 256) >> 2,  vo1 = ((tid + 256) & 3) * 8;
  const uint32_t kd0 = lds_off_of(&Ks[kr0][ko0]);
  const uint32_t kd1 = lds_off_of(&Ks[kr1][ko1]);
  const uint32_t vd0 = lds_off_of(&Vs[vr0][vo0]);
  const uint32_t vd1 = lds_off_of(&Vs[vr1][vo1]);

  v8f acc[8];
  #pragma unroll
  for (int t = 0; t < 8; ++t) acc[t] = vzero8();
  float mrow = -1e30f, lrow = 0.f;
  const float scale = 0.08838834764831845f; // 1/sqrt(128)
  const int q = q0 + l15;

  for (int s0 = 0; s0 <= smax; s0 += 32) {
    // ---- cooperative async staging of K and Vt tiles ----
    async_g2l_b128(Kr + base + (size_t)(s0 + kr0) * DK + ko0, kd0);
    async_g2l_b128(Kr + base + (size_t)(s0 + kr1) * DK + ko1, kd1);
    async_g2l_b128(Vt + vbase + (size_t)vr0 * SEQ + s0 + vo0, vd0);
    async_g2l_b128(Vt + vbase + (size_t)vr1 * SEQ + s0 + vo1, vd1);
    if (s0 + 32 <= smax)
      __builtin_prefetch(Kr + base + (size_t)(s0 + 32 + kr0) * DK + ko0, 0, 1);
    wait_async0();
    __syncthreads();

    if (s0 <= q0 + 15) {                   // this wave still inside causal range
      v8f c0 = vzero8(), c1 = vzero8();
      #pragma unroll
      for (int d = 0; d < 4; ++d) {
        union { v16bf v; v8bf h[2]; } ka;
        ka.h[0] = *(const v8bf*)&Ks[l15][32 * d + 16 * hi];
        ka.h[1] = *(const v8bf*)&Ks[l15][32 * d + 16 * hi + 8];
        c0 = wmma_bf16(ka.v, qf[d], c0);
        ka.h[0] = *(const v8bf*)&Ks[16 + l15][32 * d + 16 * hi];
        ka.h[1] = *(const v8bf*)&Ks[16 + l15][32 * d + 16 * hi + 8];
        c1 = wmma_bf16(ka.v, qf[d], c1);
      }
      // scale + causal mask + per-q tile max
      float mt = -1e30f;
      #pragma unroll
      for (int r = 0; r < 8; ++r) {
        int sA = s0 + 8 * hi + r;
        float v0 = c0[r] * scale; if (sA > q)      v0 = -1e30f; c0[r] = v0;
        float v1 = c1[r] * scale; if (sA + 16 > q) v1 = -1e30f; c1[r] = v1;
        mt = fmaxf(mt, fmaxf(v0, v1));
      }
      mt = fmaxf(mt, __shfl_xor(mt, 16, 32));
      float mnew = fmaxf(mrow, mt);
      float corr = __expf(mrow - mnew);
      float lt = 0.f;
      v8bf p0, p1;
      #pragma unroll
      for (int r = 0; r < 8; ++r) {
        float e0 = __expf(c0[r] - mnew);
        float e1 = __expf(c1[r] - mnew);
        lt += e0 + e1;
        p0[r] = (__bf16)e0;
        p1[r] = (__bf16)e1;
      }
      lt += __shfl_xor(lt, 16, 32);
      lrow = lrow * corr + lt;
      mrow = mnew;
      #pragma unroll
      for (int t = 0; t < 8; ++t)
        #pragma unroll
        for (int e = 0; e < 8; ++e) acc[t][e] *= corr;

      // P^T relayout: C frag -> LDS [q][s] -> B frag (wave-private buffer)
      *(v8bf*)&Pbuf[wv_id][l15][8 * hi]      = p0;
      *(v8bf*)&Pbuf[wv_id][l15][16 + 8 * hi] = p1;
      asm volatile("s_wait_dscnt 0" ::: "memory");
      v16bf pb = *(const v16bf*)&Pbuf[wv_id][l15][16 * hi];

      // outT += Vt(16 vdims x 32 s) x P^T(32 s x 16 q), 8 vdim tiles from LDS
      #pragma unroll
      for (int t = 0; t < 8; ++t) {
        union { v16bf v; v8bf h[2]; } va;
        va.h[0] = *(const v8bf*)&Vs[16 * t + l15][16 * hi];
        va.h[1] = *(const v8bf*)&Vs[16 * t + l15][16 * hi + 8];
        acc[t] = wmma_bf16(va.v, pb, acc[t]);
      }
    }
    __syncthreads();                       // protect LDS tiles before next stage
  }

  float rinv = 1.0f / lrow;
  int bb = bh >> 4, h = bh & 15;
  #pragma unroll
  for (int t = 0; t < 8; ++t) {
    v8bf pk;
    #pragma unroll
    for (int r = 0; r < 8; ++r) pk[r] = (__bf16)(acc[t][r] * rinv);
    __bf16* dst = Ao + ((size_t)bb * SEQ + q) * D_MODEL + h * DK + 16 * t + 8 * hi;
    *(v8bf*)dst = pk;
  }
}

// ---------------------------------------------------------------------------
// Kernel 4: output projection  out[b,s,n] = sum_f Ao[b,s,f] * Wo[n,f], f32 out
// Same async A-staging as the QKV GEMM.
// ---------------------------------------------------------------------------
__global__ void proj_o_kernel(const __bf16* __restrict__ Ao,
                              const __bf16* __restrict__ wob,
                              float* __restrict__ out) {
  __shared__ __align__(16) __bf16 As[16][136];

  const int tid  = threadIdx.x;
  const int lane = tid & 31;
  const int wv_id = tid >> 5;
  const int l15 = lane & 15;
  const int hi  = lane >> 4;

  const int NT = D_MODEL / 64;   // 32
  int task  = blockIdx.x * 8 + wv_id;   // 256*32 tasks; block shares mtile
  int ntile = task % NT;
  int mtile = task / NT;
  size_t m0 = (size_t)mtile * 16;
  int n0 = ntile * 64;

  const int arow_i = tid >> 4;
  const int achk   = (tid & 15) * 8;
  const __bf16* asrc = Ao + (m0 + (size_t)arow_i) * D_MODEL + achk;
  const uint32_t adst = lds_off_of(&As[arow_i][achk]);

  v8f acc[4];
  #pragma unroll
  for (int j = 0; j < 4; ++j) acc[j] = vzero8();

  for (int k0 = 0; k0 < D_MODEL; k0 += 128) {
    async_g2l_b128(asrc + k0, adst);
    wait_async0();
    __syncthreads();
    #pragma unroll
    for (int kk = 0; kk < 128; kk += 32) {
      union { v16bf v; v8bf h[2]; } a;
      a.h[0] = *(const v8bf*)&As[l15][kk + 16 * hi];
      a.h[1] = *(const v8bf*)&As[l15][kk + 16 * hi + 8];
      #pragma unroll
      for (int j = 0; j < 4; ++j) {
        v16bf b = *(const v16bf*)(wob + (size_t)(n0 + 16 * j + l15) * D_MODEL + k0 + kk + 16 * hi);
        acc[j] = wmma_bf16(a.v, b, acc[j]);
      }
    }
    __syncthreads();
  }

  #pragma unroll
  for (int j = 0; j < 4; ++j) {
    int n = n0 + 16 * j + l15;
    #pragma unroll
    for (int r = 0; r < 8; ++r)
      out[(m0 + r + 8 * hi) * D_MODEL + n] = acc[j][r];
  }
}

// ---------------------------------------------------------------------------
extern "C" void kernel_launch(void* const* d_in, const int* in_sizes, int n_in,
                              void* d_out, int out_size, void* d_ws, size_t ws_size,
                              hipStream_t stream) {
  const float* x  = (const float*)d_in[0];
  const float* wq = (const float*)d_in[1];
  const float* wk = (const float*)d_in[2];
  const float* wv = (const float*)d_in[3];
  const float* wo = (const float*)d_in[4];

  const size_t nX = (size_t)BATCH * SEQ * D_MODEL;   // 8388608
  const size_t nW = (size_t)D_MODEL * D_MODEL;       // 4194304
  const size_t nQ = (size_t)BATCH * NHEAD * SEQ * DK;

  char* ws = (char*)d_ws;
  size_t off = 0;
  __bf16* xb  = (__bf16*)(ws + off); off += nX * 2;
  __bf16* wqb = (__bf16*)(ws + off); off += nW * 2;
  __bf16* wkb = (__bf16*)(ws + off); off += nW * 2;
  __bf16* wvb = (__bf16*)(ws + off); off += nW * 2;
  __bf16* wob = (__bf16*)(ws + off); off += nW * 2;
  __bf16* Qr  = (__bf16*)(ws + off); off += nQ * 2;
  __bf16* Kr  = (__bf16*)(ws + off); off += nQ * 2;
  __bf16* Vt  = (__bf16*)(ws + off); off += nQ * 2;
  __bf16* Ao  = (__bf16*)(ws + off); off += nX * 2;

  cast_f32_bf16<<<(int)((nX + 255) / 256), 256, 0, stream>>>(x,  xb,  (int)nX);
  cast_f32_bf16<<<(int)((nW + 255) / 256), 256, 0, stream>>>(wq, wqb, (int)nW);
  cast_f32_bf16<<<(int)((nW + 255) / 256), 256, 0, stream>>>(wk, wkb, (int)nW);
  cast_f32_bf16<<<(int)((nW + 255) / 256), 256, 0, stream>>>(wv, wvb, (int)nW);
  cast_f32_bf16<<<(int)((nW + 255) / 256), 256, 0, stream>>>(wo, wob, (int)nW);

  // 3 mats * 256 m-tiles * 32 n-tiles = 24576 wave tasks / 8 waves = 3072 blocks
  qkv_rope_kernel<<<3072, 256, 0, stream>>>(xb, wqb, wkb, wvb, Qr, Kr, Vt);

  // 2*16*128 = 4096 wave tasks / 8 = 512 blocks
  attn_kernel<<<512, 256, 0, stream>>>(Qr, Kr, Vt, Ao);

  // 256*32 = 8192 wave tasks / 8 = 1024 blocks
  proj_o_kernel<<<1024, 256, 0, stream>>>(Ao, wob, (float*)d_out);
}
